// QuantizedLinear_79130477461727
// MI455X (gfx1250) — compile-verified, matrix operand order checked by reading, not run on hardware
//
#include <hip/hip_runtime.h>

// Problem dims
static constexpr int BDIM = 4;
static constexpr int SDIM = 2048;
static constexpr int MDIM = BDIM * SDIM;   // 8192
static constexpr int NDIM = 16384;
static constexpr int KDIM = 4096;

// GEMM tiling
static constexpr int BM = 128;             // block tile M
static constexpr int BN = 128;             // block tile N
static constexpr int KT = 64;              // K staged per LDS tile
static constexpr int LDA = KT + 8;         // LDS row stride (halves): 144B rows, 16B aligned
static constexpr int NKT = KDIM / KT;      // 64 K-iterations

typedef __attribute__((ext_vector_type(16))) _Float16 v16h;
typedef __attribute__((ext_vector_type(8)))  float    v8f;

union F16Frag { v16h v; uint4 q[2]; };
union H4      { _Float16 h[4]; uint2 q; };

// CDNA5 async global->LDS copy (GLOBAL_LOAD_ASYNC_TO_LDS_B128, ASYNCcnt-tracked).
// vdst = VGPR holding LDS byte address; vaddr = VGPR pair holding 64-bit global addr.
__device__ __forceinline__ void async_g2l_b128(unsigned int lds_addr,
                                               unsigned long long gaddr) {
    asm volatile("global_load_async_to_lds_b128 %0, %1, off"
                 :: "v"(lds_addr), "v"(gaddr) : "memory");
}

__device__ __forceinline__ void wait_async0() {
#if __has_builtin(__builtin_amdgcn_s_wait_asynccnt)
    __builtin_amdgcn_s_wait_asynccnt(0);
#else
    asm volatile("s_wait_asynccnt 0x0" ::: "memory");
#endif
}

// ---------------------------------------------------------------------------
// Prepass 1: x (fp32) -> fp16, plus per-row sums for the zero-point correction
// ---------------------------------------------------------------------------
__global__ void prep_x_kernel(const float* __restrict__ x,
                              _Float16* __restrict__ xh,
                              float* __restrict__ rsum) {
    const int m = blockIdx.x;
    const int t = threadIdx.x;
    const float4* __restrict__ xr = (const float4*)(x + (size_t)m * KDIM);
    _Float16* __restrict__ xo = xh + (size_t)m * KDIM;

    float s = 0.0f;
#pragma unroll
    for (int i = 0; i < 4; ++i) {
        const int idx = i * 256 + t;          // float4 index within the row
        float4 f = xr[idx];
        s += f.x + f.y + f.z + f.w;
        H4 h;
        h.h[0] = (_Float16)f.x; h.h[1] = (_Float16)f.y;
        h.h[2] = (_Float16)f.z; h.h[3] = (_Float16)f.w;
        *(uint2*)(xo + (size_t)idx * 4) = h.q;
    }

    __shared__ float red[256];
    red[t] = s;
    __syncthreads();
#pragma unroll
    for (int off = 128; off > 0; off >>= 1) {
        if (t < off) red[t] += red[t + off];
        __syncthreads();
    }
    if (t == 0) rsum[m] = red[0];
}

// ---------------------------------------------------------------------------
// Prepass 2: Wq (int32, int8-valued) -> fp16 (exact). 4 elements / thread.
// ---------------------------------------------------------------------------
__global__ void prep_w_kernel(const int* __restrict__ wq,
                              _Float16* __restrict__ wh) {
    const size_t idx = ((size_t)blockIdx.x * 256 + threadIdx.x) * 4;
    int4 v = *(const int4*)(wq + idx);
    H4 h;
    h.h[0] = (_Float16)v.x; h.h[1] = (_Float16)v.y;
    h.h[2] = (_Float16)v.z; h.h[3] = (_Float16)v.w;
    *(uint2*)(wh + idx) = h.q;
}

// ---------------------------------------------------------------------------
// Main GEMM: C[m,n] = scale[n] * ( sum_k xh[m,k]*wh[n,k] - zp[n]*rsum[m] )
// 256 threads = 8 waves; block tile 128x128; wave tile 64x32 (4x2 WMMA tiles)
// Double-buffered 128x64 fp16 LDS tiles filled by async global->LDS DMA.
// ---------------------------------------------------------------------------
__global__ __launch_bounds__(256)
void qlinear_wmma_kernel(const _Float16* __restrict__ xh,
                         const _Float16* __restrict__ wh,
                         const float* __restrict__ scale,
                         const int* __restrict__ zp,
                         const float* __restrict__ rsum,
                         float* __restrict__ out) {
    __shared__ _Float16 sA[2][BM * LDA];
    __shared__ _Float16 sB[2][BN * LDA];

    const int t    = threadIdx.x;
    const int lane = t & 31;
    const int wv   = t >> 5;                 // 0..7
    const int waveM = (wv >> 2) * 64;        // 0 or 64
    const int waveN = (wv & 3) * 32;         // 0,32,64,96

    const int blockM = blockIdx.y * BM;
    const int blockN = blockIdx.x * BN;

    const _Float16* __restrict__ aG = xh + (size_t)blockM * KDIM;
    const _Float16* __restrict__ bG = wh + (size_t)blockN * KDIM;

    // Per-thread chunk mapping: 1024 chunks of 8 halves per 128x64 tile.
    // 8 consecutive threads cover one row's 64 halves (128B) -> coalesced.
    const int row = (t >> 3) * 2;            // base row pattern via c = i*256+t
    (void)row;

    v8f acc[4][2];
#pragma unroll
    for (int mt = 0; mt < 4; ++mt)
#pragma unroll
        for (int nt = 0; nt < 2; ++nt)
            acc[mt][nt] = (v8f){0.f, 0.f, 0.f, 0.f, 0.f, 0.f, 0.f, 0.f};

    // Issue 8 async DMA b128 ops (4 for A, 4 for B): zero staging VGPRs.
    auto asyncCopyTile = [&](int kt, int buf) {
        const size_t kb = (size_t)kt * KT;
#pragma unroll
        for (int i = 0; i < 4; ++i) {
            const int c  = i * 256 + t;      // chunk id
            const int r  = c >> 3;           // 0..127
            const int co = (c & 7) * 8;      // half offset in row
            const unsigned int la =
                (unsigned int)(uintptr_t)&sA[buf][r * LDA + co];
            const unsigned int lb =
                (unsigned int)(uintptr_t)&sB[buf][r * LDA + co];
            async_g2l_b128(la, (unsigned long long)(uintptr_t)
                               (aG + (size_t)r * KDIM + kb + co));
            async_g2l_b128(lb, (unsigned long long)(uintptr_t)
                               (bG + (size_t)r * KDIM + kb + co));
        }
    };

    auto compute = [&](int buf) {
        const int halfK = (lane >> 4) << 3;  // 0 for lanes 0-15, 8 for lanes 16-31
        const int rsel  = lane & 15;
#pragma unroll
        for (int ks = 0; ks < KT; ks += 32) {
            F16Frag b[2];
#pragma unroll
            for (int nt = 0; nt < 2; ++nt) {
                const _Float16* p = &sB[buf][(waveN + nt * 16 + rsel) * LDA + ks + halfK];
                b[nt].q[0] = *(const uint4*)(p);
                b[nt].q[1] = *(const uint4*)(p + 16);
            }
            F16Frag a[4];
#pragma unroll
            for (int mt = 0; mt < 4; ++mt) {
                const _Float16* p = &sA[buf][(waveM + mt * 16 + rsel) * LDA + ks + halfK];
                a[mt].q[0] = *(const uint4*)(p);
                a[mt].q[1] = *(const uint4*)(p + 16);
            }
#pragma unroll
            for (int mt = 0; mt < 4; ++mt)
#pragma unroll
                for (int nt = 0; nt < 2; ++nt)
                    acc[mt][nt] = __builtin_amdgcn_wmma_f32_16x16x32_f16(
                        false, a[mt].v, false, b[nt].v,
                        (short)0, acc[mt][nt], false, false);
        }
    };

    // Prologue: DMA tile 0, wait, barrier.
    asyncCopyTile(0, 0);
    wait_async0();
    __syncthreads();

    for (int kt = 0; kt < NKT; ++kt) {
        const int buf = kt & 1;
        // Start DMA for the next tile into the other buffer (safe: every wave
        // finished reading it before the barrier that ended the last iter),
        // then crunch WMMAs on the current tile while the DMA is in flight.
        if (kt + 1 < NKT) asyncCopyTile(kt + 1, buf ^ 1);
        compute(buf);
        if (kt + 1 < NKT) {
            wait_async0();
            __syncthreads();
        }
    }

    // Epilogue: fused dequant + zero-point rank-1 correction.
    const int row0 = blockM + waveM;
    const int col0 = blockN + waveN;
    const int mhalf = (lane >> 4) << 3;      // C layout: lanes 16-31 hold rows M=8..15
    const int nsel  = lane & 15;
#pragma unroll
    for (int nt = 0; nt < 2; ++nt) {
        const int gn = col0 + nt * 16 + nsel;
        const float sc  = scale[gn];
        const float zpf = (float)zp[gn];
#pragma unroll
        for (int mt = 0; mt < 4; ++mt) {
#pragma unroll
            for (int r = 0; r < 8; ++r) {
                const int gm = row0 + mt * 16 + mhalf + r;
                out[(size_t)gm * NDIM + gn] = sc * (acc[mt][nt][r] - zpf * rsum[gm]);
            }
        }
    }
}

// ---------------------------------------------------------------------------
extern "C" void kernel_launch(void* const* d_in, const int* in_sizes, int n_in,
                              void* d_out, int out_size, void* d_ws, size_t ws_size,
                              hipStream_t stream) {
    (void)in_sizes; (void)n_in; (void)out_size; (void)ws_size;

    const float* x     = (const float*)d_in[0];
    const int*   wq    = (const int*)d_in[1];
    const float* scale = (const float*)d_in[2];
    const int*   zp    = (const int*)d_in[3];
    float*       out   = (float*)d_out;

    // Workspace layout: xh (fp16, 64MB) | wh (fp16, 128MB) | rsum (fp32, 32KB)
    char* ws = (char*)d_ws;
    _Float16* xh   = (_Float16*)ws;
    _Float16* wh   = (_Float16*)(ws + (size_t)MDIM * KDIM * sizeof(_Float16));
    float*    rsum = (float*)(ws + (size_t)MDIM * KDIM * sizeof(_Float16)
                                 + (size_t)NDIM * KDIM * sizeof(_Float16));

    prep_x_kernel<<<MDIM, 256, 0, stream>>>(x, xh, rsum);

    const size_t wElems = (size_t)NDIM * KDIM;           // 67,108,864
    prep_w_kernel<<<(int)(wElems / (256 * 4)), 256, 0, stream>>>(wq, wh);

    dim3 grid(NDIM / BN, MDIM / BM);                     // (128, 64)
    qlinear_wmma_kernel<<<grid, 256, 0, stream>>>(xh, wh, scale, zp, rsum, out);
}
